// VideoCLIPAttention_15006615733411
// MI455X (gfx1250) — compile-verified, matrix operand order checked by reading, not run on hardware
//
#include <hip/hip_runtime.h>

#define EMBED 512
#define NHEADS 8
#define HDIM 64
#define BATCH 4
#define SEQ 2048
#define QK_SCALE 0.125f  // 64^-0.5

typedef __attribute__((ext_vector_type(16))) __bf16 v16bf;
typedef __attribute__((ext_vector_type(8)))  __bf16 v8bf;
typedef __attribute__((ext_vector_type(8)))  float  v8f;

// ---------------------------------------------------------------------------
// WMMA fragment helpers (CDNA5 16x16x32 bf16, wave32).
// A-frag (16x32, MxK): lane row = lane&15; per-lane elements cover
//   K = {half*8 .. half*8+7} and {16+half*8 .. 16+half*8+7}, half = lane>>4.
// B-frag (32x16, KxN): identical addressing with "row" = column N.
// C/D frag (16x16 f32): VGPR r holds row M = r + 8*(lane>>4), col N = lane&15.
// ---------------------------------------------------------------------------
__device__ __forceinline__ v16bf load_frag(const __bf16* base, int stride, int lane) {
    const __bf16* p = base + (lane & 15) * stride + ((lane >> 4) & 1) * 8;
    union { v16bf v; v8bf h[2]; } u;
    u.h[0] = *(const v8bf*)(p);
    u.h[1] = *(const v8bf*)(p + 16);
    return u.v;
}

__device__ __forceinline__ v8f wmma_bf16(v16bf a, v16bf b, v8f c) {
    return __builtin_amdgcn_wmma_f32_16x16x32_bf16(
        false, a, false, b, (short)0, c, false, false);
}

// DPP16 lane-permute (pure VALU, no DS pipe). CTRL: 0xB1 quad_perm xor1,
// 0x4E quad_perm xor2, 0x141 row_half_mirror (^7), 0x140 row_mirror (^15).
template <int CTRL>
__device__ __forceinline__ float dpp_movf(float x) {
    return __builtin_bit_cast(float,
        __builtin_amdgcn_update_dpp(0, __builtin_bit_cast(int, x),
                                    CTRL, 0xF, 0xF, true));
}

// Butterfly max over the 16-lane half-wave group (xor1,2,7,15 closure).
__device__ __forceinline__ float redmax16(float v) {
    v = fmaxf(v, dpp_movf<0xB1>(v));
    v = fmaxf(v, dpp_movf<0x4E>(v));
    v = fmaxf(v, dpp_movf<0x141>(v));
    v = fmaxf(v, dpp_movf<0x140>(v));
    return v;
}

__device__ __forceinline__ float redsum16(float v) {
    v += dpp_movf<0xB1>(v);
    v += dpp_movf<0x4E>(v);
    v += dpp_movf<0x141>(v);
    v += dpp_movf<0x140>(v);
    return v;
}

// ---------------------------------------------------------------------------
// fp32 -> bf16 elementwise (weights)
// ---------------------------------------------------------------------------
__global__ void cvt_f32_bf16(const float* __restrict__ in, __bf16* __restrict__ out, int n) {
    int i = blockIdx.x * blockDim.x + threadIdx.x;
    if (i < n) out[i] = (__bf16)in[i];
}

// ---------------------------------------------------------------------------
// GEMM: out[8192 x 512] = X[8192 x 512] @ W^T[512 x 512] + bias
// Block = 256 threads (8 waves), each block does a 16-row x 512-col strip.
// ---------------------------------------------------------------------------
template <typename TIN, bool HEADS>
__global__ __launch_bounds__(256) void proj_kernel(const TIN* __restrict__ X,
                                                   const __bf16* __restrict__ W,
                                                   const float* __restrict__ bias,
                                                   void* __restrict__ outp) {
    __shared__ __align__(16) __bf16 As[16 * 520];
    const int tid  = threadIdx.x;
    const int row0 = blockIdx.x * 16;

    for (int i = tid; i < 16 * EMBED; i += 256) {
        int r = i >> 9, c = i & 511;
        As[r * 520 + c] = (__bf16)X[(row0 + r) * EMBED + c];
    }
    __syncthreads();

    const int wave = tid >> 5, lane = tid & 31;
    const int colbase = wave * 64;
    v8f acc[4] = {};

#pragma unroll 4
    for (int ks = 0; ks < 16; ++ks) {
        v16bf a = load_frag(As + ks * 32, 520, lane);
#pragma unroll
        for (int t = 0; t < 4; ++t) {
            v16bf bf = load_frag(W + (colbase + t * 16) * EMBED + ks * 32, EMBED, lane);
            acc[t] = wmma_bf16(a, bf, acc[t]);
        }
    }

    const int g = lane >> 4, n = lane & 15;
#pragma unroll
    for (int t = 0; t < 4; ++t) {
        int col = colbase + t * 16 + n;
        float bv = bias[col];
#pragma unroll
        for (int r = 0; r < 8; ++r) {
            int row = row0 + r + 8 * g;
            float val = acc[t][r] + bv;
            if (HEADS) {
                int b = row >> 11, nn = row & (SEQ - 1);
                int h = col >> 6,  d  = col & (HDIM - 1);
                ((__bf16*)outp)[(((size_t)(b * NHEADS + h) * SEQ) + nn) * HDIM + d] = (__bf16)val;
            } else {
                ((float*)outp)[(size_t)row * EMBED + col] = val;
            }
        }
    }
}

// ---------------------------------------------------------------------------
// Flash attention: grid.x = B*H, grid.y = N/128. 8 waves x 16 query rows.
// K tile staged via async global->LDS (ASYNCcnt); V transposed via VGPRs.
// Row max reduced per tile with DPP16; row sum kept as per-lane partials
// (rescale factor is row-uniform) and reduced once in the epilogue.
// ---------------------------------------------------------------------------
__global__ __launch_bounds__(256) void attn_kernel(const __bf16* __restrict__ qh,
                                                   const __bf16* __restrict__ kh,
                                                   const __bf16* __restrict__ vh,
                                                   __bf16* __restrict__ xo) {
    __shared__ __align__(16) __bf16 Ks[32 * 72];       // K tile [key][d]
    __shared__ __align__(16) __bf16 Vt[64 * 40];       // V tile transposed [d][key]
    __shared__ __align__(16) __bf16 Ps[8 * 16 * 40];   // per-wave P staging [row][key]

    const int tid = threadIdx.x, wave = tid >> 5, lane = tid & 31;
    const int bh = blockIdx.x;
    const int b = bh >> 3, h = bh & 7;
    const __bf16* Q = qh + (size_t)bh * SEQ * HDIM;
    const __bf16* K = kh + (size_t)bh * SEQ * HDIM;
    const __bf16* V = vh + (size_t)bh * SEQ * HDIM;
    const int qrow0 = blockIdx.y * 128 + wave * 16;

    v16bf qa0 = load_frag(Q + qrow0 * HDIM, HDIM, lane);       // d 0..31
    v16bf qa1 = load_frag(Q + qrow0 * HDIM + 32, HDIM, lane);  // d 32..63
    __bf16* Pw = Ps + wave * (16 * 40);

    float m[8], l[8];
#pragma unroll
    for (int r = 0; r < 8; ++r) { m[r] = -3.0e38f; l[r] = 0.f; }
    v8f o0 = {}, o1 = {}, o2 = {}, o3 = {};

    const int e0 = tid * 8;            // 256 threads x 8 elems = 32x64 tile
    const int kr = e0 >> 6, kc = e0 & 63;
    const unsigned ks_lds = (unsigned)(uintptr_t)(Ks + kr * 72 + kc);

    for (int j = 0; j < SEQ; j += 32) {
        __syncthreads();  // previous tile fully consumed
        {
            // K tile: async global -> LDS, one b128 per thread (4 KB total).
            unsigned long long ga =
                (unsigned long long)(uintptr_t)(K + (j + kr) * HDIM + kc);
            asm volatile("global_load_async_to_lds_b128 %0, %1, off"
                         :: "v"(ks_lds), "v"(ga) : "memory");
            // V tile transposed through VGPRs (2B-granular scatter).
            v8bf vv = *(const v8bf*)(V + (j + kr) * HDIM + kc);
#pragma unroll
            for (int t = 0; t < 8; ++t) Vt[(kc + t) * 40 + kr] = vv[t];
        }
        asm volatile("s_wait_asynccnt 0x0" ::: "memory");
        __syncthreads();

        // S = Q K^T : two 16x16 tiles (keys j..j+15, j+16..j+31), K-dim = 64.
        v8f s0 = {}, s1 = {};
        {
            v16bf b00 = load_frag(Ks, 72, lane);
            v16bf b01 = load_frag(Ks + 32, 72, lane);
            v16bf b10 = load_frag(Ks + 16 * 72, 72, lane);
            v16bf b11 = load_frag(Ks + 16 * 72 + 32, 72, lane);
            s0 = wmma_bf16(qa0, b00, s0);
            s0 = wmma_bf16(qa1, b01, s0);
            s1 = wmma_bf16(qa0, b10, s1);
            s1 = wmma_bf16(qa1, b11, s1);
        }

        // Online softmax update (DPP max reduce; per-lane partial row sums).
        float sc[8];
#pragma unroll
        for (int r = 0; r < 8; ++r) {
            float v0 = s0[r] * QK_SCALE;
            float v1 = s1[r] * QK_SCALE;
            float mx = redmax16(fmaxf(v0, v1));
            float mn = fmaxf(m[r], mx);
            float p0 = __expf(v0 - mn);
            float p1 = __expf(v1 - mn);
            float f  = __expf(m[r] - mn);
            l[r] = l[r] * f + p0 + p1;   // partial over this lane's 2 columns
            m[r] = mn;
            sc[r] = f;
            int prow = r + 8 * (lane >> 4);  // C-layout row -> A-layout staging
            Pw[prow * 40 + (lane & 15)]      = (__bf16)p0;
            Pw[prow * 40 + 16 + (lane & 15)] = (__bf16)p1;
        }
#pragma unroll
        for (int r = 0; r < 8; ++r) {
            o0[r] *= sc[r]; o1[r] *= sc[r]; o2[r] *= sc[r]; o3[r] *= sc[r];
        }

        // Wave-local LDS RAW on Pw: wait DS ops before fragment reload.
        asm volatile("s_wait_dscnt 0x0" ::: "memory");

        v16bf pa  = load_frag(Pw, 40, lane);            // P 16x32 A-frag
        v16bf vb0 = load_frag(Vt, 40, lane);            // V^T cols d 0..15
        v16bf vb1 = load_frag(Vt + 16 * 40, 40, lane);
        v16bf vb2 = load_frag(Vt + 32 * 40, 40, lane);
        v16bf vb3 = load_frag(Vt + 48 * 40, 40, lane);
        o0 = wmma_bf16(pa, vb0, o0);
        o1 = wmma_bf16(pa, vb1, o1);
        o2 = wmma_bf16(pa, vb2, o2);
        o3 = wmma_bf16(pa, vb3, o3);
    }

    // Epilogue: finish the row-sum reduction once, normalize, write bf16
    // activations [B,N,E] (head-interleaved).
    const int g = lane >> 4, n = lane & 15;
#pragma unroll
    for (int r = 0; r < 8; ++r) {
        float inv = 1.0f / redsum16(l[r]);
        int row = qrow0 + r + 8 * g;
        size_t base = ((size_t)b * SEQ + row) * EMBED + h * HDIM;
        xo[base + 0 + n]  = (__bf16)(o0[r] * inv);
        xo[base + 16 + n] = (__bf16)(o1[r] * inv);
        xo[base + 32 + n] = (__bf16)(o2[r] * inv);
        xo[base + 48 + n] = (__bf16)(o3[r] * inv);
    }
}

// ---------------------------------------------------------------------------
// Launch: cvt weights -> Q/K/V projections -> flash attention -> out proj.
// Workspace: 3x headed QKV + attn activations (8.39MB each) + bf16 weights
// (~2MB) = ~35.6 MB.
// ---------------------------------------------------------------------------
extern "C" void kernel_launch(void* const* d_in, const int* in_sizes, int n_in,
                              void* d_out, int out_size, void* d_ws, size_t ws_size,
                              hipStream_t stream) {
    const float* query = (const float*)d_in[0];
    const float* key   = (const float*)d_in[1];
    const float* value = (const float*)d_in[2];
    const float* Wq = (const float*)d_in[3]; const float* bq = (const float*)d_in[4];
    const float* Wk = (const float*)d_in[5]; const float* bk = (const float*)d_in[6];
    const float* Wv = (const float*)d_in[7]; const float* bv = (const float*)d_in[8];
    const float* Wo = (const float*)d_in[9]; const float* bo = (const float*)d_in[10];

    char* ws = (char*)d_ws;
    const size_t sz_h = (size_t)BATCH * NHEADS * SEQ * HDIM * sizeof(__bf16);
    __bf16* qh  = (__bf16*)(ws + 0 * sz_h);
    __bf16* kh  = (__bf16*)(ws + 1 * sz_h);
    __bf16* vh  = (__bf16*)(ws + 2 * sz_h);
    __bf16* xa  = (__bf16*)(ws + 3 * sz_h);
    __bf16* wqb = (__bf16*)(ws + 4 * sz_h);
    __bf16* wkb = wqb + EMBED * EMBED;
    __bf16* wvb = wkb + EMBED * EMBED;
    __bf16* wob = wvb + EMBED * EMBED;

    const int wn = EMBED * EMBED;
    cvt_f32_bf16<<<(wn + 255) / 256, 256, 0, stream>>>(Wq, wqb, wn);
    cvt_f32_bf16<<<(wn + 255) / 256, 256, 0, stream>>>(Wk, wkb, wn);
    cvt_f32_bf16<<<(wn + 255) / 256, 256, 0, stream>>>(Wv, wvb, wn);
    cvt_f32_bf16<<<(wn + 255) / 256, 256, 0, stream>>>(Wo, wob, wn);

    const int rowtiles = (BATCH * SEQ) / 16;  // 512
    proj_kernel<float, true><<<rowtiles, 256, 0, stream>>>(query, wqb, bq, qh);
    proj_kernel<float, true><<<rowtiles, 256, 0, stream>>>(key,   wkb, bk, kh);
    proj_kernel<float, true><<<rowtiles, 256, 0, stream>>>(value, wvb, bv, vh);

    attn_kernel<<<dim3(BATCH * NHEADS, SEQ / 128), 256, 0, stream>>>(qh, kh, vh, xa);

    proj_kernel<__bf16, false><<<rowtiles, 256, 0, stream>>>(xa, wob, bo, d_out);
}
